// HRGRAgent_27522150432853
// MI455X (gfx1250) — compile-verified, matrix operand order checked by reading, not run on hardware
//
#include <hip/hip_runtime.h>
#include <hip/hip_bf16.h>
#include <math.h>

// ---------------- problem constants ----------------
#define Bsz   64
#define Nsp   49
#define ENC   1024
#define DEC   512
#define EMB   512
#define VOC   10000
#define NT    4
#define MS    8
#define MW    20
#define STEPS (MS * MW)          // 160
#define HROWS (STEPS * Bsz)      // 10240

// ---------------- vector types for WMMA ----------------
typedef __attribute__((ext_vector_type(16))) __bf16 v16bf;
typedef __attribute__((ext_vector_type(8)))  __bf16 v8bf;
typedef __attribute__((ext_vector_type(8)))  float  v8f;

__device__ __forceinline__ __bf16 f2bf(float f) {
  union { float f; unsigned u; } v; v.f = f;
  unsigned r = v.u + 0x7FFFu + ((v.u >> 16) & 1u);   // RNE
  unsigned short h = (unsigned short)(r >> 16);
  return __builtin_bit_cast(__bf16, h);
}

// Load one 16x32 bf16 WMMA fragment (A or B) for this lane.
// p points at (row m|n of the matrix) + current k offset; rows are K-contiguous.
// Per ISA 7.12.2: elems 0..7 at k = 8*half, elems 8..15 at k = 16 + 8*half.
__device__ __forceinline__ v16bf ldfrag(const __bf16* p, int half) {
  v8bf lo = *(const v8bf*)(p + (half << 3));
  v8bf hi = *(const v8bf*)(p + 16 + (half << 3));
  v16bf r;
#pragma unroll
  for (int i = 0; i < 8; ++i) { r[i] = lo[i]; r[i + 8] = hi[i]; }
  return r;
}

#define WMMA_BF16(Af, Bf, Cf) \
  __builtin_amdgcn_wmma_f32_16x16x32_bf16(false, (Af), false, (Bf), (short)0, (Cf), false, false)

// ---------------------------------------------------------------------------
// Generic WMMA GEMM:  C[M,N] (f32 row-major) = A[M,K](bf16) * W[N,K](bf16)^T + bias[N]
// block = 256 threads = 8 waves; wave w -> 16-row tile (blockIdx.y*8 + w),
// 4 x 16-col tiles starting at blockIdx.x*4 (A fragment reused across 4 WMMAs).
// Invalid column tiles (only in the last N-block) are CLAMPED to the last
// valid tile so the k-loop is branch-free; only the store is guarded.
// fc_remap: C row r=(step*64+b) is stored at output row (b*160+step), giving
// the [B, MS, MW, V] layout directly.
// ---------------------------------------------------------------------------
__global__ void __launch_bounds__(256) wmma_gemm_bias(
    const __bf16* __restrict__ A, const __bf16* __restrict__ W,
    const float* __restrict__ bias, float* __restrict__ C,
    int M, int N, int K, int fc_remap)
{
  const int lane = threadIdx.x & 31;
  const int wave = threadIdx.x >> 5;
  const int tile_m = blockIdx.y * 8 + wave;
  const int row0 = tile_m << 4;
  if (row0 >= M) return;                      // wave-uniform: EXEC stays full
  const int n_tiles = N >> 4;
  const int ct0 = blockIdx.x * 4;
  const int mn = lane & 15;
  const int half = lane >> 4;

  // clamp out-of-range tiles to a safe, valid tile (loads harmless, store skipped)
  const int c0 = (ct0 + 0 < n_tiles) ? (ct0 + 0) : (n_tiles - 1);
  const int c1 = (ct0 + 1 < n_tiles) ? (ct0 + 1) : (n_tiles - 1);
  const int c2 = (ct0 + 2 < n_tiles) ? (ct0 + 2) : (n_tiles - 1);
  const int c3 = (ct0 + 3 < n_tiles) ? (ct0 + 3) : (n_tiles - 1);

  const __bf16* arow = A + (size_t)(row0 + mn) * K;
  const __bf16* b0 = W + (size_t)((c0 << 4) + mn) * K;
  const __bf16* b1 = W + (size_t)((c1 << 4) + mn) * K;
  const __bf16* b2 = W + (size_t)((c2 << 4) + mn) * K;
  const __bf16* b3 = W + (size_t)((c3 << 4) + mn) * K;

  v8f acc0 = {0.f,0.f,0.f,0.f,0.f,0.f,0.f,0.f};
  v8f acc1 = acc0, acc2 = acc0, acc3 = acc0;

  for (int kk = 0; kk < K; kk += 32) {
    v16bf af = ldfrag(arow + kk, half);
    v16bf f0 = ldfrag(b0 + kk, half);
    v16bf f1 = ldfrag(b1 + kk, half);
    v16bf f2 = ldfrag(b2 + kk, half);
    v16bf f3 = ldfrag(b3 + kk, half);
    acc0 = WMMA_BF16(af, f0, acc0);
    acc1 = WMMA_BF16(af, f1, acc1);
    acc2 = WMMA_BF16(af, f2, acc2);
    acc3 = WMMA_BF16(af, f3, acc3);
  }

  // store: D element (r, lane&15) holds C[row0 + r + 8*half][ct*16 + lane&15]
  v8f accs[4] = {acc0, acc1, acc2, acc3};
#pragma unroll
  for (int t = 0; t < 4; ++t) {
    if (ct0 + t >= n_tiles) break;            // block-uniform
    const int col = ((ct0 + t) << 4) + mn;
    const float bv = bias[col];
#pragma unroll
    for (int r = 0; r < 8; ++r) {
      int row = row0 + r + (half << 3);
      int orow = fc_remap ? ((row & 63) * 160 + (row >> 6)) : row;
      C[(size_t)orow * N + col] = accs[t][r] + bv;
    }
  }
}

// ---------------- prep kernels ----------------
__global__ void f32_to_bf16_kernel(const float* __restrict__ in,
                                   __bf16* __restrict__ out, int n) {
  for (int i = blockIdx.x * blockDim.x + threadIdx.x; i < n;
       i += gridDim.x * blockDim.x)
    out[i] = f2bf(in[i]);
}

__global__ void global_mean_kernel(const float* __restrict__ spatial,
                                   float* __restrict__ gf) {
  int b = blockIdx.x;
  for (int c = threadIdx.x; c < ENC; c += blockDim.x) {
    float s = 0.f;
#pragma unroll 7
    for (int n = 0; n < Nsp; ++n) s += spatial[((size_t)b * Nsp + n) * ENC + c];
    gf[b * ENC + c] = s * (1.0f / 49.0f);
  }
}

__global__ void init_h_kernel(const float* __restrict__ gf,
                              const float* __restrict__ initW,
                              const float* __restrict__ initb,
                              float* __restrict__ h_s) {
  int b = blockIdx.x, d = threadIdx.x;           // block = 512
  const float* g = gf + b * ENC;
  const float* w = initW + (size_t)d * ENC;
  float s = initb[d];
#pragma unroll 8
  for (int k = 0; k < ENC; ++k) s += g[k] * w[k];
  h_s[b * DEC + d] = tanhf(s);
}

// gi for the sentence GRU is step-invariant (input is always global features).
__global__ void gi_s_kernel(const float* __restrict__ gf,
                            const float* __restrict__ sWih,
                            const float* __restrict__ sbih,
                            float* __restrict__ gi_s) {
  int b = blockIdx.x;
  int j = blockIdx.y * DEC + threadIdx.x;        // grid (64,3), block 512
  const float* g = gf + b * ENC;
  const float* w = sWih + (size_t)j * ENC;
  float s = sbih[j];
#pragma unroll 8
  for (int k = 0; k < ENC; ++k) s += g[k] * w[k];
  gi_s[b * (3 * DEC) + j] = s;
}

// ---------------- sequential chain ----------------
__global__ void sent_gru_kernel(const float* __restrict__ gi_s,
                                const float* __restrict__ sWhh,
                                const float* __restrict__ sbhh,
                                float* __restrict__ h_s,
                                float* __restrict__ h_w) {
  __shared__ float h[DEC];
  int b = blockIdx.x, d = threadIdx.x;           // block = 512
  h[d] = h_s[b * DEC + d];
  __syncthreads();
  float hr = sbhh[d], hz = sbhh[DEC + d], hn = sbhh[2 * DEC + d];
  const float* wr = sWhh + (size_t)d * DEC;
  const float* wz = sWhh + (size_t)(DEC + d) * DEC;
  const float* wn = sWhh + (size_t)(2 * DEC + d) * DEC;
#pragma unroll 4
  for (int k = 0; k < DEC; ++k) {
    float hv = h[k];
    hr += wr[k] * hv; hz += wz[k] * hv; hn += wn[k] * hv;
  }
  const float* gi = gi_s + b * (3 * DEC);
  float r = 1.f / (1.f + expf(-(gi[d] + hr)));
  float z = 1.f / (1.f + expf(-(gi[DEC + d] + hz)));
  float n = tanhf(gi[2 * DEC + d] + r * hn);
  float hnew = (1.f - z) * n + z * h[d];
  h_s[b * DEC + d] = hnew;
  h_w[b * DEC + d] = hnew;                       // word-scan initial carry
}

__global__ void policy_stop_kernel(const float* __restrict__ h_s,
                                   const float* __restrict__ pW,
                                   const float* __restrict__ pb,
                                   const float* __restrict__ stW,
                                   const float* __restrict__ stb,
                                   float* __restrict__ out_policy,
                                   float* __restrict__ out_stop, int s) {
  int b = blockIdx.x;                            // block = 192 (6 waves)
  int lane = threadIdx.x & 31, j = threadIdx.x >> 5;
  const float* h = h_s + b * DEC;
  const float* w = (j < 5) ? (pW + (size_t)j * DEC) : stW;
  float acc = 0.f;
  for (int k = lane; k < DEC; k += 32) acc += h[k] * w[k];
#pragma unroll
  for (int off = 16; off > 0; off >>= 1) acc += __shfl_down(acc, off, 32);
  if (lane == 0) {
    if (j < 5) out_policy[(b * MS + s) * (1 + NT) + j] = acc + pb[j];
    else       out_stop[b * MS + s] = acc + stb[0];
  }
}

__global__ void att_ctx_kernel(const float* __restrict__ h_w,
                               const float* __restrict__ attdecW,
                               const float* __restrict__ attdecb,
                               const float* __restrict__ att1,
                               const float* __restrict__ attfullW,
                               const float* __restrict__ attfullb,
                               const float* __restrict__ spatial,
                               float* __restrict__ ctx) {
  __shared__ float h[DEC];
  __shared__ float a2[EMB];
  __shared__ float e[Nsp];
  int b = blockIdx.x, tid = threadIdx.x;         // block = 256
  for (int i = tid; i < DEC; i += 256) h[i] = h_w[b * DEC + i];
  __syncthreads();
  for (int j = tid; j < EMB; j += 256) {
    const float* w = attdecW + (size_t)j * DEC;
    float s = attdecb[j];
#pragma unroll 4
    for (int k = 0; k < DEC; ++k) s += h[k] * w[k];
    a2[j] = s;
  }
  __syncthreads();
  int lane = tid & 31, wv = tid >> 5;
  for (int n = wv; n < Nsp; n += 8) {
    const float* a1 = att1 + ((size_t)(b * Nsp + n)) * EMB;
    float s = 0.f;
    for (int k = lane; k < EMB; k += 32) {
      float v = a1[k] + a2[k];
      v = v > 0.f ? v : 0.f;
      s += v * attfullW[k];
    }
#pragma unroll
    for (int off = 16; off > 0; off >>= 1) s += __shfl_down(s, off, 32);
    if (lane == 0) e[n] = s + attfullb[0];
  }
  __syncthreads();
  if (tid == 0) {                                // softmax over 49 (tiny)
    float mx = e[0];
    for (int n = 1; n < Nsp; ++n) mx = fmaxf(mx, e[n]);
    float sum = 0.f;
    for (int n = 0; n < Nsp; ++n) { float ex = expf(e[n] - mx); e[n] = ex; sum += ex; }
    float inv = 1.f / sum;
    for (int n = 0; n < Nsp; ++n) e[n] *= inv;
  }
  __syncthreads();
  for (int c = tid; c < ENC; c += 256) {
    float s = 0.f;
#pragma unroll 7
    for (int n = 0; n < Nsp; ++n)
      s += e[n] * spatial[((size_t)(b * Nsp + n)) * ENC + c];
    ctx[b * ENC + c] = s;
  }
}

__global__ void word_gru_kernel(const int* __restrict__ target,
                                const float* __restrict__ embW,
                                const float* __restrict__ ctx,
                                const float* __restrict__ wWih,
                                const float* __restrict__ wWhh,
                                const float* __restrict__ wbih,
                                const float* __restrict__ wbhh,
                                float* __restrict__ h_w,
                                __bf16* __restrict__ Hbf, int s, int t) {
  __shared__ float x[EMB + ENC];                 // 1536 floats
  __shared__ float h[DEC];
  int b = blockIdx.x, d = threadIdx.x;           // block = 512
  int idx = (t == 0) ? 1 : target[(b * MS + s) * MW + (t - 1)];
  x[d]             = embW[(size_t)idx * EMB + d];
  x[EMB + d]       = ctx[b * ENC + d];
  x[EMB + DEC + d] = ctx[b * ENC + DEC + d];
  h[d] = h_w[b * DEC + d];
  __syncthreads();

  const int XK = EMB + ENC;                      // 1536
  float ir = wbih[d], iz = wbih[DEC + d], in_ = wbih[2 * DEC + d];
  const float* wr = wWih + (size_t)d * XK;
  const float* wz = wWih + (size_t)(DEC + d) * XK;
  const float* wn = wWih + (size_t)(2 * DEC + d) * XK;
#pragma unroll 4
  for (int k = 0; k < XK; ++k) {
    float xv = x[k];
    ir += wr[k] * xv; iz += wz[k] * xv; in_ += wn[k] * xv;
  }
  float hr = wbhh[d], hz = wbhh[DEC + d], hn = wbhh[2 * DEC + d];
  const float* vr = wWhh + (size_t)d * DEC;
  const float* vz = wWhh + (size_t)(DEC + d) * DEC;
  const float* vn = wWhh + (size_t)(2 * DEC + d) * DEC;
#pragma unroll 4
  for (int k = 0; k < DEC; ++k) {
    float hv = h[k];
    hr += vr[k] * hv; hz += vz[k] * hv; hn += vn[k] * hv;
  }
  float r = 1.f / (1.f + expf(-(ir + hr)));
  float z = 1.f / (1.f + expf(-(iz + hz)));
  float n = tanhf(in_ + r * hn);
  float hnew = (1.f - z) * n + z * h[d];
  h_w[b * DEC + d] = hnew;
  int step = s * MW + t;
  Hbf[((size_t)step * Bsz + b) * DEC + d] = f2bf(hnew);
}

// ---------------------------------------------------------------------------
extern "C" void kernel_launch(void* const* d_in, const int* in_sizes, int n_in,
                              void* d_out, int out_size, void* d_ws, size_t ws_size,
                              hipStream_t stream) {
  const float* spatial    = (const float*)d_in[0];
  const int*   target     = (const int*)  d_in[1];
  const float* emb_W      = (const float*)d_in[2];
  const float* s_W_ih     = (const float*)d_in[3];
  const float* s_W_hh     = (const float*)d_in[4];
  const float* s_b_ih     = (const float*)d_in[5];
  const float* s_b_hh     = (const float*)d_in[6];
  const float* policy_W   = (const float*)d_in[7];
  const float* policy_b   = (const float*)d_in[8];
  const float* stop_W     = (const float*)d_in[9];
  const float* stop_b     = (const float*)d_in[10];
  const float* att_enc_W  = (const float*)d_in[11];
  const float* att_enc_b  = (const float*)d_in[12];
  const float* att_dec_W  = (const float*)d_in[13];
  const float* att_dec_b  = (const float*)d_in[14];
  const float* att_full_W = (const float*)d_in[15];
  const float* att_full_b = (const float*)d_in[16];
  const float* w_W_ih     = (const float*)d_in[17];
  const float* w_W_hh     = (const float*)d_in[18];
  const float* w_b_ih     = (const float*)d_in[19];
  const float* w_b_hh     = (const float*)d_in[20];
  const float* fc_W       = (const float*)d_in[21];
  const float* fc_b       = (const float*)d_in[22];
  const float* init_W     = (const float*)d_in[23];
  const float* init_b     = (const float*)d_in[24];

  float* out        = (float*)d_out;
  float* out_policy = out;                         // [64,8,5]
  float* out_stop   = out + Bsz * MS * (1 + NT);   // [64,8,1]
  float* out_words  = out_stop + Bsz * MS;         // [64,8,20,10000]

  // ---- carve workspace (~36 MB) ----
  char* w = (char*)d_ws;
  auto carve = [&](size_t bytes) -> void* {
    void* p = (void*)w;
    w += (bytes + 255) & ~(size_t)255;
    return p;
  };
  __bf16* spatial_bf = (__bf16*)carve((size_t)Bsz * Nsp * ENC * 2);
  __bf16* attencW_bf = (__bf16*)carve((size_t)EMB * ENC * 2);
  __bf16* fcW_bf     = (__bf16*)carve((size_t)VOC * DEC * 2);
  float*  att1       = (float*) carve((size_t)Bsz * Nsp * EMB * 4);
  float*  gf         = (float*) carve((size_t)Bsz * ENC * 4);
  float*  gi_s       = (float*) carve((size_t)Bsz * 3 * DEC * 4);
  float*  h_s        = (float*) carve((size_t)Bsz * DEC * 4);
  float*  h_w        = (float*) carve((size_t)Bsz * DEC * 4);
  float*  ctx        = (float*) carve((size_t)Bsz * ENC * 4);
  __bf16* Hbf        = (__bf16*)carve((size_t)HROWS * DEC * 2);

  // ---- prep: bf16 conversions + step-invariant precompute ----
  f32_to_bf16_kernel<<<1024, 256, 0, stream>>>(spatial, spatial_bf, Bsz * Nsp * ENC);
  f32_to_bf16_kernel<<<1024, 256, 0, stream>>>(att_enc_W, attencW_bf, EMB * ENC);
  f32_to_bf16_kernel<<<2048, 256, 0, stream>>>(fc_W, fcW_bf, VOC * DEC);
  global_mean_kernel<<<Bsz, 256, 0, stream>>>(spatial, gf);
  init_h_kernel<<<Bsz, DEC, 0, stream>>>(gf, init_W, init_b, h_s);
  gi_s_kernel<<<dim3(Bsz, 3), DEC, 0, stream>>>(gf, s_W_ih, s_b_ih, gi_s);

  // att1[3136,512] = spatial[3136,1024] x att_enc_W^T  (WMMA)
  // M tiles: 196 -> grid.y = 25; N tiles: 32 -> grid.x = 8
  wmma_gemm_bias<<<dim3(8, 25), 256, 0, stream>>>(
      spatial_bf, attencW_bf, att_enc_b, att1, Bsz * Nsp, EMB, ENC, 0);

  // ---- sequential recurrence (thin, latency-bound) ----
  for (int s = 0; s < MS; ++s) {
    sent_gru_kernel<<<Bsz, DEC, 0, stream>>>(gi_s, s_W_hh, s_b_hh, h_s, h_w);
    policy_stop_kernel<<<Bsz, 192, 0, stream>>>(h_s, policy_W, policy_b,
                                                stop_W, stop_b,
                                                out_policy, out_stop, s);
    for (int t = 0; t < MW; ++t) {
      att_ctx_kernel<<<Bsz, 256, 0, stream>>>(h_w, att_dec_W, att_dec_b, att1,
                                              att_full_W, att_full_b, spatial, ctx);
      word_gru_kernel<<<Bsz, DEC, 0, stream>>>(target, emb_W, ctx,
                                               w_W_ih, w_W_hh, w_b_ih, w_b_hh,
                                               h_w, Hbf, s, t);
    }
  }

  // ---- batched vocab projection (dominant GEMM, WMMA) ----
  // C[10240,10000] = H[10240,512] x fc_W[10000,512]^T + fc_b,
  // rows permuted into [B, MS, MW, V] on store.
  // N tiles: 625 -> grid.x = ceil(625/4) = 157; M tiles: 640 -> grid.y = 80.
  wmma_gemm_bias<<<dim3(157, 80), 256, 0, stream>>>(
      Hbf, fcW_bf, fc_b, out_words, HROWS, VOC, DEC, 1);
}
